// AttentionSequencePoolingLayer_29042568855621
// MI455X (gfx1250) — matrix operand order repaired and checked by reading, not compiled, not run on hardware
//
#include <hip/hip_runtime.h>
#include <hip/hip_bf16.h>

typedef __attribute__((ext_vector_type(16))) _Float16 v16h;
typedef __attribute__((ext_vector_type(8)))  _Float16 v8h;
typedef __attribute__((ext_vector_type(8)))  float    v8f;

#define BATCH   2048
#define T_LEN   200
#define E_DIM   64
#define H1_DIM  80
#define H2_DIM  40
#define KT_STR  68      // keys-tile LDS row stride in floats (bank-conflict-free, 16B aligned)
#define H1_PAD  96      // h1 K-dim padded to 3 x 32
#define N_TILES1 5      // 80 / 16
#define K_STEPS1 8      // 256 / 32
#define N_TILES2 3      // 48 / 16
#define K_STEPS2 3      // 96 / 32
#define FRAG_HALFS 512  // 32 lanes * 16 halves
#define W1_HALFS (K_STEPS1 * N_TILES1 * FRAG_HALFS)   // 20480
#define W2_HALFS (K_STEPS2 * N_TILES2 * FRAG_HALFS)   // 4608
#define WF_HALFS (W1_HALFS + W2_HALFS)                // 25088 -> 50176 B

// ---------------------------------------------------------------------------
// Kernel 0: pack W1/W2 (f32) into f16 WMMA B-fragment layout in workspace.
// B fragment (32x16 KxN, f16): lane = hi*16 + n, element e -> K = hi*8 + e + (e>=8 ? 8 : 0).
// Fragment f stored at f*512 + lane*16 + e.  w2 frags follow w1 frags.
// ---------------------------------------------------------------------------
__global__ void pack_weights_kernel(const float* __restrict__ W1,
                                    const float* __restrict__ W2,
                                    _Float16* __restrict__ wf) {
    int gid = blockIdx.x * blockDim.x + threadIdx.x;
    if (gid < W1_HALFS) {
        int f    = gid >> 9;
        int r    = gid & 511;
        int lane = r >> 4;
        int e    = r & 15;
        int kk   = f / N_TILES1;
        int nt   = f % N_TILES1;
        int hi   = lane >> 4;
        int K    = kk * 32 + hi * 8 + e + (e >= 8 ? 8 : 0);          // < 256 always
        int N    = nt * 16 + (lane & 15);          // < 80 always
        wf[gid] = (_Float16)W1[K * H1_DIM + N];
    } else if (gid < WF_HALFS) {
        int g    = gid - W1_HALFS;
        int f    = g >> 9;
        int r    = g & 511;
        int lane = r >> 4;
        int e    = r & 15;
        int kk   = f / N_TILES2;
        int nt   = f % N_TILES2;
        int hi   = lane >> 4;
        int K    = kk * 32 + hi * 8 + e + (e >= 8 ? 8 : 0);          // padded: >= 80 -> 0
        int N    = nt * 16 + (lane & 15);          // padded: >= 40 -> 0
        float v  = (K < H1_DIM && N < H2_DIM) ? W2[K * H2_DIM + N] : 0.0f;
        wf[gid]  = (_Float16)v;
    }
}

// ---------------------------------------------------------------------------
// Main fused kernel: one workgroup per batch, 8 wave32s, WMMA MLP + pooling.
// All WMMA B operands are served from LDS (weights staged once per block).
// ---------------------------------------------------------------------------
__global__ __launch_bounds__(256) void din_attention_pool_kernel(
        const float* __restrict__ query,       // [B,1,64]
        const float* __restrict__ keys,        // [B,200,64]
        const int*   __restrict__ keys_length, // [B,1]
        const float* __restrict__ b1,          // [80]
        const float* __restrict__ b2,          // [40]
        const float* __restrict__ W3,          // [40,1]
        const float* __restrict__ b3,          // [1]
        const _Float16* __restrict__ wf,       // packed B frags (w1 || w2)
        float* __restrict__ out)               // [B,1,64]
{
    __shared__ __align__(16) _Float16 wfs[WF_HALFS];   // 50176 B weight frags
    __shared__ __align__(16) float    qs[E_DIM];
    __shared__ float    out_acc[E_DIM];
    __shared__ __align__(16) float    kt[8][16][KT_STR];   // per-wave keys tile
    __shared__ __align__(16) _Float16 h1s[8][16][H1_PAD];  // per-wave hidden tile
    __shared__ float    scores_s[8][16];                   // per-wave tile scores

    const int b    = blockIdx.x;
    const int tid  = threadIdx.x;
    const int wave = tid >> 5;
    const int lane = tid & 31;
    const int hi   = lane >> 4;
    const int lx   = lane & 15;

    // ---- stage packed weight fragments into LDS (contiguous b128 copies) ----
    {
        const uint4* src = (const uint4*)wf;
        uint4*       dst = (uint4*)wfs;
        const int n4 = WF_HALFS / 8;                 // 3136 uint4
        for (int i = tid; i < n4; i += 256) dst[i] = src[i];
    }
    if (tid < E_DIM) {
        qs[tid]      = query[(size_t)b * E_DIM + tid];
        out_acc[tid] = 0.0f;
    }
    __syncthreads();

    const int klen = keys_length[b];
    const float b3v = b3[0];

    // Per-lane bias / W3 values for the N column this lane owns.
    float b1v[N_TILES1];
#pragma unroll
    for (int nt = 0; nt < N_TILES1; ++nt) b1v[nt] = b1[nt * 16 + lx];
    float b2v[N_TILES2], w3v[N_TILES2];
#pragma unroll
    for (int nt = 0; nt < N_TILES2; ++nt) {
        int N = nt * 16 + lx;
        b2v[nt] = (N < H2_DIM) ? b2[N] : 0.0f;
        w3v[nt] = (N < H2_DIM) ? W3[N] : 0.0f;
    }

    // 13 M-tiles of 16 timesteps each (200 -> padded 208), round-robin by wave.
    for (int tile = wave; tile < 13; tile += 8) {
        // Keep weight-fragment LDS loads inside the loop (no LICM hoist ->
        // no >256-VGPR pressure, no scratch spills around the WMMAs).
        asm volatile("" ::: "memory");

        const int t0 = tile * 16;

        // ---- stage 16x64 keys tile into LDS (clamped rows; masked later) ----
#pragma unroll
        for (int i = 0; i < 8; ++i) {
            int idx = lane + 32 * i;           // 0..255 -> 16 rows x 16 float4
            int row = idx >> 4;
            int c4  = idx & 15;
            int t   = t0 + row;
            if (t >= T_LEN) t = T_LEN - 1;
            const float4 v = *(const float4*)(keys +
                ((size_t)b * T_LEN + t) * E_DIM + c4 * 4);
            *(float4*)&kt[wave][row][c4 * 4] = v;
        }

        // ---------------- GEMM1: [16x256] x [256x80] --------------------
        v8f acc1[N_TILES1];
#pragma unroll
        for (int nt = 0; nt < N_TILES1; ++nt)
            acc1[nt] = (v8f){0.f,0.f,0.f,0.f,0.f,0.f,0.f,0.f};

#pragma unroll
        for (int kk = 0; kk < K_STEPS1; ++kk) {
            // A-fragment K mapping: K_local = hi*8 + e + (e>=8 ? 8 : 0)
            // feature column c = (K & 63); runs: c0..c0+7 and c0+16..c0+23
            const int seg = kk >> 1;                 // 0:q 1:k 2:q-k 3:q*k
            const int c0  = (kk & 1) * 32 + hi * 8;
            float qv[16], kv[16];
            *(float4*)&qv[0]  = *(const float4*)&qs[c0];
            *(float4*)&qv[4]  = *(const float4*)&qs[c0 + 4];
            *(float4*)&qv[8]  = *(const float4*)&qs[c0 + 16];
            *(float4*)&qv[12] = *(const float4*)&qs[c0 + 20];
            *(float4*)&kv[0]  = *(const float4*)&kt[wave][lx][c0];
            *(float4*)&kv[4]  = *(const float4*)&kt[wave][lx][c0 + 4];
            *(float4*)&kv[8]  = *(const float4*)&kt[wave][lx][c0 + 16];
            *(float4*)&kv[12] = *(const float4*)&kt[wave][lx][c0 + 20];

            v16h a;
#pragma unroll
            for (int e = 0; e < 16; ++e) {
                float f;
                if      (seg == 0) f = qv[e];
                else if (seg == 1) f = kv[e];
                else if (seg == 2) f = qv[e] - kv[e];
                else               f = qv[e] * kv[e];
                a[e] = (_Float16)f;
            }
#pragma unroll
            for (int nt = 0; nt < N_TILES1; ++nt) {
                const v16h bf = *(const v16h*)&wfs[(kk * N_TILES1 + nt) * FRAG_HALFS + lane * 16];
                acc1[nt] = __builtin_amdgcn_wmma_f32_16x16x32_f16(
                    false, a, false, bf, (short)0, acc1[nt], false, false);
            }
        }

        // ---- bias + sigmoid, store h1 tile (f16) to LDS, zero K-pad ----
#pragma unroll
        for (int nt = 0; nt < N_TILES1; ++nt)
#pragma unroll
            for (int r = 0; r < 8; ++r) {
                float x = acc1[nt][r] + b1v[nt];
                float s = 1.0f / (1.0f + __expf(-x));
                h1s[wave][r + hi * 8][nt * 16 + lx] = (_Float16)s;
            }
#pragma unroll
        for (int i = 0; i < 8; ++i) {            // pad cols 80..95 with zeros
            int idx = lane + 32 * i;             // 256 pad elements
            h1s[wave][idx >> 4][H1_DIM + (idx & 15)] = (_Float16)0.0f;
        }

        // ---------------- GEMM2: [16x96] x [96x48] ----------------------
        v8f acc2[N_TILES2];
#pragma unroll
        for (int nt = 0; nt < N_TILES2; ++nt)
            acc2[nt] = (v8f){0.f,0.f,0.f,0.f,0.f,0.f,0.f,0.f};

#pragma unroll
        for (int kk = 0; kk < K_STEPS2; ++kk) {
            const int c0 = kk * 32 + hi * 8;
            const v8h alo = *(const v8h*)&h1s[wave][lx][c0];
            const v8h ahi = *(const v8h*)&h1s[wave][lx][c0 + 16];
            v16h a;
#pragma unroll
            for (int e = 0; e < 8; ++e) { a[e] = alo[e]; a[e + 8] = ahi[e]; }
#pragma unroll
            for (int nt = 0; nt < N_TILES2; ++nt) {
                const v16h bf = *(const v16h*)&wfs[W1_HALFS + (kk * N_TILES2 + nt) * FRAG_HALFS + lane * 16];
                acc2[nt] = __builtin_amdgcn_wmma_f32_16x16x32_f16(
                    false, a, false, bf, (short)0, acc2[nt], false, false);
            }
        }

        // ---- sigmoid + W3 dot + cross-lane reduce + length mask ----
#pragma unroll
        for (int r = 0; r < 8; ++r) {
            float s = 0.0f;
#pragma unroll
            for (int nt = 0; nt < N_TILES2; ++nt) {
                float x = acc2[nt][r] + b2v[nt];
                float h = 1.0f / (1.0f + __expf(-x));
                s += h * w3v[nt];                 // w3v==0 masks pad columns
            }
            s += __shfl_xor(s, 8, 16);
            s += __shfl_xor(s, 4, 16);
            s += __shfl_xor(s, 2, 16);
            s += __shfl_xor(s, 1, 16);
            if (lx == 0) {
                int m  = r + hi * 8;
                int t  = t0 + m;
                float sc = s + b3v;
                if (t >= T_LEN || t >= klen) sc = 0.0f;
                scores_s[wave][m] = sc;
            }
        }

        // ---- pooling: out_acc[e] += sum_t score[t] * keys[t][e] ----
#pragma unroll
        for (int half = 0; half < 2; ++half) {
            int e = lane + 32 * half;
            float p = 0.0f;
#pragma unroll
            for (int j = 0; j < 16; ++j)
                p += scores_s[wave][j] * kt[wave][j][e];
            atomicAdd(&out_acc[e], p);
        }
    }

    __syncthreads();
    if (tid < E_DIM) out[(size_t)b * E_DIM + tid] = out_acc[tid];
}

// ---------------------------------------------------------------------------
extern "C" void kernel_launch(void* const* d_in, const int* in_sizes, int n_in,
                              void* d_out, int out_size, void* d_ws, size_t ws_size,
                              hipStream_t stream) {
    const float* query  = (const float*)d_in[0];
    const float* keys   = (const float*)d_in[1];
    const int*   klen   = (const int*)  d_in[2];
    const float* W1     = (const float*)d_in[3];
    const float* b1     = (const float*)d_in[4];
    const float* W2     = (const float*)d_in[5];
    const float* b2     = (const float*)d_in[6];
    const float* W3     = (const float*)d_in[7];
    const float* b3     = (const float*)d_in[8];
    float* out          = (float*)d_out;

    _Float16* wf = (_Float16*)d_ws;          // w1 frags || w2 frags, ~49 KB

    pack_weights_kernel<<<(WF_HALFS + 255) / 256, 256, 0, stream>>>(W1, W2, wf);

    din_attention_pool_kernel<<<BATCH, 256, 0, stream>>>(
        query, keys, klen, b1, b2, W3, b3, wf, out);
}
